// SecondOrderScorer_44659069943962
// MI455X (gfx1250) — matrix operand order repaired and checked by reading, not compiled, not run on hardware
//
#include <hip/hip_runtime.h>
#include <hip/hip_bf16.h>

// B=2, L=128, D_IN=1024, D_TRI=128, N_OUT=2
typedef __bf16 bf16;
typedef __attribute__((ext_vector_type(16))) __bf16 v16bf;
typedef __attribute__((ext_vector_type(8)))  __bf16 v8bf;
typedef __attribute__((ext_vector_type(8)))  float   v8f;
typedef __attribute__((address_space(3))) char lds_char;

union AB16 { v16bf v; v8bf h[2]; };

// ---- fragment loaders (row-major sources; B supplied as B^T row-major) ----
// 16-bit A-matrix 16x32: lane l holds row (l&15); VGPR0-3 = K (l>>4)*8 + 0..7,
// VGPR4-7 = K 16 + (l>>4)*8 + 0..7  -> two contiguous 16B chunks.
__device__ __forceinline__ v16bf load_frag_a(const bf16* base, int ld, int row0,
                                             int k0, int lane) {
  int r  = row0 + (lane & 15);
  int kb = k0 + ((lane >> 4) << 3);
  const bf16* p = base + r * ld + kb;
  AB16 f;
  f.h[0] = *(const v8bf*)(p);
  f.h[1] = *(const v8bf*)(p + 16);
  return f.v;
}

// 16-bit B-matrix 32x16: lane l holds column (l&15); K = (l>>4)*8 + 0..7,
// then K 16 + (l>>4)*8 + 0..7 -> two 16B chunks from B^T row (l&15).
__device__ __forceinline__ v16bf load_frag_b(const bf16* base, int ld, int n0,
                                             int k0, int lane) {
  int n  = n0 + (lane & 15);
  int kb = k0 + ((lane >> 4) << 3);
  const bf16* p = base + n * ld + kb;
  AB16 f;
  f.h[0] = *(const v8bf*)(p);
  f.h[1] = *(const v8bf*)(p + 16);
  return f.v;
}

__device__ __forceinline__ v8f wmma_bf16(v16bf a, v16bf b, v8f c) {
  return __builtin_amdgcn_wmma_f32_16x16x32_bf16(false, a, false, b,
                                                 (short)0, c, false, false);
}

// async copy of 16B per lane: global -> LDS, tracked by ASYNCcnt
__device__ __forceinline__ void async_g2l_b128(const void* gaddr, void* laddr) {
  unsigned long long ga = (unsigned long long)gaddr;          // flat==global
  unsigned la = (unsigned)(unsigned long long)(lds_char*)laddr; // LDS byte addr
  asm volatile("global_load_async_to_lds_b128 %0, %1, off"
               :: "v"(la), "v"(ga) : "memory");
}
__device__ __forceinline__ void wait_async0() {
  asm volatile("s_wait_asynccnt 0x0" ::: "memory");
}

// ---------------- conversion kernels ----------------
__global__ void cvt_f32_bf16(const float* __restrict__ in, bf16* __restrict__ out,
                             int n) {
  int i = blockIdx.x * blockDim.x + threadIdx.x;
  if (i < n) out[i] = (bf16)in[i];
}

// mlp_W [8][1024][128] -> [8][128][1024] (B^T) bf16
__global__ void cvt_mlpw(const float* __restrict__ in, bf16* __restrict__ out) {
  int i = blockIdx.x * blockDim.x + threadIdx.x;   // 8*128*1024 = 1048576
  if (i >= 8 * 128 * 1024) return;
  int m = i >> 17, r = i & 131071;
  int n = r >> 10, k = r & 1023;
  out[i] = (bf16)in[(m << 17) + k * 128 + n];
}

// tri_W [16(to)][i][k][j] -> [16][j*128+i][k] bf16  (B^T for wz GEMM)
__global__ void cvt_triw(const float* __restrict__ in, bf16* __restrict__ out) {
  const long total = 16L * 128 * 128 * 128;        // 33.5M
  long stride = (long)gridDim.x * blockDim.x;
  for (long idx = blockIdx.x * (long)blockDim.x + threadIdx.x; idx < total;
       idx += stride) {
    int to = (int)(idx >> 21);
    long r = idx & 2097151;
    int j = (int)(r >> 14), ii = (int)((r >> 7) & 127), k = (int)(r & 127);
    out[idx] = (bf16)in[((long)to << 21) + ((long)ii * 128 + k) * 128 + j];
  }
}

// ---------------- MLP projections: proj[m][bz][d] bf16 ----------------
__global__ __launch_bounds__(256) void mlp_kernel(const bf16* __restrict__ xb,
                                                  const bf16* __restrict__ wt,
                                                  const float* __restrict__ bias,
                                                  bf16* __restrict__ proj) {
  int lane = threadIdx.x & 31;
  int wl = blockIdx.x * 8 + (threadIdx.x >> 5);    // 1024 tiles
  int m = wl >> 7, rem = wl & 127;
  int mt = rem >> 3, nt = rem & 7;
  const bf16* Bt = wt + m * 128 * 1024;
  v8f acc = {};
  for (int k0 = 0; k0 < 1024; k0 += 32) {
    v16bf a = load_frag_a(xb, 1024, mt * 16, k0, lane);
    v16bf b = load_frag_b(Bt, 1024, nt * 16, k0, lane);
    acc = wmma_bf16(a, b, acc);
  }
  int N = nt * 16 + (lane & 15);
  float bv = bias[m * 128 + N];
  int mhi = (lane >> 4) << 3;
#pragma unroll
  for (int v = 0; v < 8; v++) {
    int M = mt * 16 + v + mhi;
    float val = acc[v] + bv;
    val = val > 0.f ? val : 0.1f * val;              // LeakyReLU(0.1)
    proj[(m * 256 + M) * 128 + N] = (bf16)val;
  }
}

// -------- wz stage: wzT[b][o][z][j*128+i] = sum_k z[bz,k] * w[o,i,k,j] --------
__global__ __launch_bounds__(256) void wz_kernel(const bf16* __restrict__ projz,
                                                 const bf16* __restrict__ wt_t,
                                                 bf16* __restrict__ wz) {
  int lane = threadIdx.x & 31;
  int wl = blockIdx.x * 8 + (threadIdx.x >> 5);    // 2*16*1024 = 32768 tiles
  int o = wl >> 14, rem = wl & 16383;
  int mt = rem >> 10, nt = rem & 1023;
  const bf16* Bt = wt_t + o * (16384 * 128);
  v8f acc = {};
#pragma unroll
  for (int k0 = 0; k0 < 128; k0 += 32) {
    v16bf a = load_frag_a(projz, 128, mt * 16, k0, lane);
    v16bf b = load_frag_b(Bt, 128, nt * 16, k0, lane);
    acc = wmma_bf16(a, b, acc);
  }
  int N = nt * 16 + (lane & 15);
  int mhi = (lane >> 4) << 3;
#pragma unroll
  for (int v = 0; v < 8; v++) {
    int M = mt * 16 + v + mhi;                      // M = bz
    int b_ = M >> 7, z = M & 127;
    wz[(((b_ * 2 + o) * 128 + z) * 16384) + N] = (bf16)acc[v];
  }
}

// -------- fused s & out: one block per (b,o,z); two 128^3 GEMMs + epilogue ----
// mode 0: plain, 1: sym_triu, 2: sym_triu + cop transpose ([b,x,y,z,o])
__global__ __launch_bounds__(256) void tri_kernel(const bf16* __restrict__ proj,
                                                  const bf16* __restrict__ wz,
                                                  float* __restrict__ out,
                                                  int xm, int ym, int mode) {
  __shared__ __align__(16) bf16 s_stage[16384];    // wz tile, 32 KiB
  extern __shared__ char smem[];
  bf16*  s_lds = (bf16*)smem;                      // [128][136] bf16
  float* c_lds = (float*)(smem + 128 * 136 * 2);   // [128][130] f32
  int lane = threadIdx.x & 31;
  int wave = threadIdx.x >> 5;                     // x row-strip 0..7
  int z = blockIdx.x & 127;
  int o = (blockIdx.x >> 7) & 1;
  int b = blockIdx.x >> 8;

  // kick off async DMA of the per-block wz tile (32 KiB) into LDS
  const bf16* B1g = wz + (((b * 2 + o) * 128 + z) * 16384);
  {
    const char* gsrc = (const char*)B1g;
    char* ldst = (char*)s_stage;
#pragma unroll
    for (int k = 0; k < 8; k++) {
      int off = (threadIdx.x + k * 256) * 16;      // 2048 x 16B chunks
      async_g2l_b128(gsrc + off, ldst + off);
    }
  }

  // GEMM1 A fragments (overlap with the async DMA)
  const bf16* A1 = proj + (xm * 256 + b * 128) * 128;
  v16bf afr[4];
#pragma unroll
  for (int kk = 0; kk < 4; kk++) afr[kk] = load_frag_a(A1, 128, wave * 16, kk * 32, lane);
  int mhi = (lane >> 4) << 3;

  wait_async0();
  __syncthreads();

  // GEMM1: s[x,j] = sum_i xproj[x,i] * wz[i,j]   (B^T = wzT[j][i] from LDS)
#pragma unroll
  for (int nt = 0; nt < 8; nt++) {
    v8f acc = {};
#pragma unroll
    for (int kk = 0; kk < 4; kk++)
      acc = wmma_bf16(afr[kk], load_frag_b(s_stage, 128, nt * 16, kk * 32, lane), acc);
    int N = nt * 16 + (lane & 15);
#pragma unroll
    for (int v = 0; v < 8; v++)
      s_lds[(wave * 16 + v + mhi) * 136 + N] = (bf16)acc[v];
  }
  __syncthreads();

  // GEMM2: out[x,y] = sum_j s[x,j] * yproj[y,j]   (B^T = yproj row-major)
  const bf16* B2 = proj + (ym * 256 + b * 128) * 128;
  v16bf af2[4];
#pragma unroll
  for (int kk = 0; kk < 4; kk++)
    af2[kk] = load_frag_a(s_lds, 136, wave * 16, kk * 32, lane);
#pragma unroll
  for (int nt = 0; nt < 8; nt++) {
    v8f acc = {};
#pragma unroll
    for (int kk = 0; kk < 4; kk++)
      acc = wmma_bf16(af2[kk], load_frag_b(B2, 128, nt * 16, kk * 32, lane), acc);
    int N = nt * 16 + (lane & 15);
#pragma unroll
    for (int v = 0; v < 8; v++)
      c_lds[(wave * 16 + v + mhi) * 130 + N] = acc[v];
  }
  __syncthreads();

  // epilogue: sym_triu + layout mapping, fp32 writes
#pragma unroll 4
  for (int i = 0; i < 64; i++) {
    int e = threadIdx.x + i * 256;
    int x = e >> 7, y = e & 127;
    float val = (mode >= 1 && x > y) ? c_lds[y * 130 + x] : c_lds[x * 130 + y];
    long idx = (mode == 2)
                   ? ((((long)b * 128 + x) * 128 + y) * 128 + z) * 2 + o
                   : ((((long)b * 128 + z) * 128 + x) * 128 + y) * 2 + o;
    out[idx] = val;
  }
}

extern "C" void kernel_launch(void* const* d_in, const int* in_sizes, int n_in,
                              void* d_out, int out_size, void* d_ws, size_t ws_size,
                              hipStream_t stream) {
  const float* x     = (const float*)d_in[0];   // [2,128,1024]
  const float* mlp_W = (const float*)d_in[1];   // [8,1024,128]
  const float* mlp_b = (const float*)d_in[2];   // [8,128]
  const float* tri_W = (const float*)d_in[3];   // [8,2,128,128,128]
  float* out = (float*)d_out;

  char* ws = (char*)d_ws;
  bf16* x_bf   = (bf16*)(ws);                         // 256K elems (512 KiB)
  bf16* proj   = (bf16*)(ws + (1u << 19));            // 8*256*128   (512 KiB)
  bf16* mlpWt  = (bf16*)(ws + (1u << 20));            // 8*128*1024  (2 MiB)
  bf16* triWt  = (bf16*)(ws + (3u << 20));            // 16*128^3    (64 MiB)
  bf16* wz     = (bf16*)(ws + (67u << 20));           // 2*2*128*16384 (16 MiB)

  cvt_f32_bf16<<<(262144 + 255) / 256, 256, 0, stream>>>(x, x_bf, 262144);
  cvt_mlpw<<<(1048576 + 255) / 256, 256, 0, stream>>>(mlp_W, mlpWt);
  cvt_triw<<<8192, 256, 0, stream>>>(tri_W, triWt);
  mlp_kernel<<<128, 256, 0, stream>>>(x_bf, mlpWt, mlp_b, proj);

  // per-triaffine config (tri_W order) -> output slot (return order), roles
  static const int t_slot[8] = {1, 0, 3, 2, 4, 5, 6, 7};
  static const int t_mode[8] = {0, 0, 0, 0, 1, 1, 2, 2};
  static const int t_xm[8]   = {4, 4, 1, 1, 4, 5, 0, 1};  // rh rh pt pt rh rt ph pt
  static const int t_ym[8]   = {5, 5, 5, 4, 4, 5, 0, 1};  // rt rt rt rh rh rt ph pt
  static const int t_zm[8]   = {1, 0, 0, 0, 0, 1, 4, 5};  // pt ph ph ph ph pt rh rt

  const size_t slot_elems = 2L * 128 * 128 * 128 * 2;     // 8388608
  const size_t lds_bytes = 128 * 136 * 2 + 128 * 130 * 4; // 101376 dynamic

  for (int t = 0; t < 8; t++) {
    wz_kernel<<<4096, 256, 0, stream>>>(proj + t_zm[t] * 256 * 128,
                                        triWt + (size_t)t * 2 * 16384 * 128, wz);
    tri_kernel<<<512, 256, lds_bytes, stream>>>(proj, wz,
                                                out + t_slot[t] * slot_elems,
                                                t_xm[t], t_ym[t], t_mode[t]);
  }
}
